// SRALayer_53506702573936
// MI455X (gfx1250) — matrix operand order repaired and checked by reading, not compiled, not run on hardware
//
#include <hip/hip_runtime.h>
#include <hip/hip_bf16.h>

typedef __attribute__((ext_vector_type(16))) _Float16 v16h;
typedef __attribute__((ext_vector_type(8)))  float    v8f;
typedef __attribute__((ext_vector_type(4)))  float    f32x4;

#define DDIM  512
#define SLEN  2048
#define BATCH 8
#define MTOT  (BATCH * SLEN)   // 16384
#define EB    32               // scan column-block width

// ---------------- f32 -> f16 convert ----------------
__global__ void cvt_f16_kernel(const float* __restrict__ in,
                               _Float16* __restrict__ out, int n) {
  int i = blockIdx.x * blockDim.x + threadIdx.x;
  if (i < n) out[i] = (_Float16)in[i];
}

// ------- W (N x K, f32) -> Wt (K x N, f16), 512x512 -------
__global__ void transpose_f16_kernel(const float* __restrict__ W,
                                     _Float16* __restrict__ Wt) {
  int idx = blockIdx.x * blockDim.x + threadIdx.x;   // coalesced writes
  int k = idx >> 9;
  int n = idx & (DDIM - 1);
  Wt[idx] = (_Float16)W[n * DDIM + k];
}

// ---------------- WMMA GEMM: C = A @ Wt + bias (opt. sigmoid) ----------------
// A : M x 512 row-major f16, Wt : 512 x 512 (K-major) f16, C : M x 512 f32.
// One wave computes a 64x64 tile (4x4 WMMA sub-tiles); 8 waves per block.
__global__ void __launch_bounds__(256)
gemm_f16_kernel(const _Float16* __restrict__ A,
                const _Float16* __restrict__ Bt,
                const float* __restrict__ bias,
                float* __restrict__ C,
                int M, int act)
{
  const int wave = threadIdx.x >> 5;
  const int lane = threadIdx.x & 31;
  const int tile = blockIdx.x * 8 + wave;
  const int tilesPerRow = DDIM / 64;             // 8
  const int m0 = (tile / tilesPerRow) * 64;
  const int n0 = (tile % tilesPerRow) * 64;
  if (m0 >= M) return;

  // A fragment addressing (16-bit A 16x32 layout):
  //   lane<16 : row +lane,    K chunks [k0..k0+7]  and [k0+16..k0+23]
  //   lane>=16: row +lane-16, K chunks [k0+8..15] and [k0+24..31]
  const int mrow  = m0 + (lane & 15);
  const int khalf = (lane >> 4) * 8;

  v8f acc[4][4];
#pragma unroll
  for (int mi = 0; mi < 4; ++mi)
#pragma unroll
    for (int nt = 0; nt < 4; ++nt) acc[mi][nt] = (v8f){};

  for (int k0 = 0; k0 < DDIM; k0 += 32) {
    v16h a[4], b[4];
#pragma unroll
    for (int mi = 0; mi < 4; ++mi) {
      const _Float16* ap = A + (size_t)(mrow + 16 * mi) * DDIM + k0 + khalf;
      ((f32x4*)&a[mi])[0] = *(const f32x4*)(ap);
      ((f32x4*)&a[mi])[1] = *(const f32x4*)(ap + 16);
    }
    // B fragment (32x16): lane holds column n0+lane%16, K in same order as A.
    const _Float16* bp = Bt + (size_t)(k0 + khalf) * DDIM + n0 + (lane & 15);
#pragma unroll
    for (int nt = 0; nt < 4; ++nt) {
#pragma unroll
      for (int s = 0; s < 16; ++s)
        b[nt][s] = bp[(size_t)(s < 8 ? s : s + 8) * DDIM + nt * 16];
    }
#pragma unroll
    for (int mi = 0; mi < 4; ++mi)
#pragma unroll
      for (int nt = 0; nt < 4; ++nt)
        acc[mi][nt] = __builtin_amdgcn_wmma_f32_16x16x32_f16(
            false, a[mi], false, b[nt], (short)0, acc[mi][nt], false, false);
  }

  // Epilogue: sub-tile C layout -> VGPR i holds M = +i + (lane<16 ? 0 : 8),
  // N = +lane%16
  const int nc     = lane & 15;
  const int mshift = (lane >> 4) * 8;
#pragma unroll
  for (int mi = 0; mi < 4; ++mi) {
#pragma unroll
    for (int nt = 0; nt < 4; ++nt) {
      const int n = n0 + nt * 16 + nc;
      const float bv = bias[n];
#pragma unroll
      for (int i = 0; i < 8; ++i) {
        float v = acc[mi][nt][i] + bv;
        if (act) v = __builtin_amdgcn_rcpf(1.0f + __expf(-v));   // fast sigmoid
        C[(size_t)(m0 + 16 * mi + mshift + i) * DDIM + n] = v;
      }
    }
  }
}

// ---------------- recurrent scan ----------------
// grid: BATCH * (DDIM/EB) blocks, 256 threads (8 waves).
// Wave w owns state rows d in [w*64, w*64+64); lane owns column e0+lane.
// State slab 512 x 32 f32 lives in registers (64 floats / thread).
__global__ void __launch_bounds__(256)
scan_kernel(const float* __restrict__ Q, const float* __restrict__ K,
            const float* __restrict__ V, const float* __restrict__ G,
            float* __restrict__ O)
{
  const int nEB  = DDIM / EB;                  // 16
  const int b    = blockIdx.x / nEB;
  const int e0   = (blockIdx.x % nEB) * EB;
  const int w    = threadIdx.x >> 5;
  const int lane = threadIdx.x & 31;

  __shared__ __align__(16) float sh_q[DDIM];
  __shared__ __align__(16) float sh_k[DDIM];
  __shared__ __align__(16) float sh_g[DDIM];
  __shared__ float sh_v[EB];
  __shared__ float partial[8][EB + 1];

  float st[64];
#pragma unroll
  for (int j = 0; j < 64; ++j) st[j] = 0.0f;

  const size_t bbase = (size_t)b * SLEN * DDIM;
  const float* qb = Q + bbase;
  const float* kb = K + bbase;
  const float* vb = V + bbase;
  const float* gb = G + bbase;

  const int tid2 = threadIdx.x * 2;

  for (int t = 0; t < SLEN; ++t) {
    const size_t base = (size_t)t * DDIM;

    // stage q/k/g rows (512 f32 each) + v slice into LDS, coalesced
    *(float2*)(sh_q + tid2) = *(const float2*)(qb + base + tid2);
    *(float2*)(sh_k + tid2) = *(const float2*)(kb + base + tid2);
    *(float2*)(sh_g + tid2) = *(const float2*)(gb + base + tid2);
    if (threadIdx.x < EB) sh_v[threadIdx.x] = vb[base + e0 + threadIdx.x];
    if (t + 1 < SLEN) {   // prefetch next timestep (global_prefetch_b8)
      __builtin_prefetch(qb + base + DDIM + tid2, 0, 1);
      __builtin_prefetch(kb + base + DDIM + tid2, 0, 1);
      __builtin_prefetch(gb + base + DDIM + tid2, 0, 1);
    }
    __syncthreads();

    const float v = sh_v[lane];
    float acc = 0.0f;
    const f32x4* q4 = (const f32x4*)(sh_q + (w << 6));
    const f32x4* k4 = (const f32x4*)(sh_k + (w << 6));
    const f32x4* g4 = (const f32x4*)(sh_g + (w << 6));
#pragma unroll
    for (int jj = 0; jj < 16; ++jj) {
      const f32x4 g = g4[jj];
      const f32x4 k = k4[jj];
      const f32x4 q = q4[jj];
      const int j = jj * 4;
      st[j + 0] = fmaf(g.x, st[j + 0], k.x * v); acc = fmaf(q.x, st[j + 0], acc);
      st[j + 1] = fmaf(g.y, st[j + 1], k.y * v); acc = fmaf(q.y, st[j + 1], acc);
      st[j + 2] = fmaf(g.z, st[j + 2], k.z * v); acc = fmaf(q.z, st[j + 2], acc);
      st[j + 3] = fmaf(g.w, st[j + 3], k.w * v); acc = fmaf(q.w, st[j + 3], acc);
    }
    partial[w][lane] = acc;
    __syncthreads();

    if (w == 0) {
      float s = 0.0f;
#pragma unroll
      for (int ww = 0; ww < 8; ++ww) s += partial[ww][lane];
      O[bbase + base + e0 + lane] = s;
    }
  }
}

// ---------------- launcher ----------------
extern "C" void kernel_launch(void* const* d_in, const int* in_sizes, int n_in,
                              void* d_out, int out_size, void* d_ws, size_t ws_size,
                              hipStream_t stream) {
  const float* x  = (const float*)d_in[0];
  const float* Wq = (const float*)d_in[1];  const float* bq = (const float*)d_in[2];
  const float* Wk = (const float*)d_in[3];  const float* bk = (const float*)d_in[4];
  const float* Wv = (const float*)d_in[5];  const float* bv = (const float*)d_in[6];
  const float* Wg = (const float*)d_in[7];  const float* bg = (const float*)d_in[8];
  const float* Wo = (const float*)d_in[9];  const float* bo = (const float*)d_in[10];
  float* out = (float*)d_out;

  char* ws = (char*)d_ws;
  size_t off = 0;
  auto carve = [&](size_t bytes) -> void* {
    void* p = ws + off;
    off = (off + bytes + 255) & ~(size_t)255;
    return p;
  };

  const size_t MD = (size_t)MTOT * DDIM;
  _Float16* xb   = (_Float16*)carve(MD * sizeof(_Float16));
  _Float16* Ob   = (_Float16*)carve(MD * sizeof(_Float16));
  _Float16* Wtq  = (_Float16*)carve((size_t)DDIM * DDIM * sizeof(_Float16));
  _Float16* Wtk  = (_Float16*)carve((size_t)DDIM * DDIM * sizeof(_Float16));
  _Float16* Wtv  = (_Float16*)carve((size_t)DDIM * DDIM * sizeof(_Float16));
  _Float16* Wtg  = (_Float16*)carve((size_t)DDIM * DDIM * sizeof(_Float16));
  _Float16* Wto  = (_Float16*)carve((size_t)DDIM * DDIM * sizeof(_Float16));
  float* Qb = (float*)carve(MD * sizeof(float));
  float* Kb = (float*)carve(MD * sizeof(float));
  float* Vb = (float*)carve(MD * sizeof(float));
  float* Gb = (float*)carve(MD * sizeof(float));
  float* Oi = (float*)carve(MD * sizeof(float));
  (void)ws_size; (void)in_sizes; (void)n_in; (void)out_size;

  // 1) convert x to f16
  {
    int n = (int)MD;
    cvt_f16_kernel<<<(n + 255) / 256, 256, 0, stream>>>(x, xb, n);
  }
  // 2) transpose+convert weights (K-major f16)
  {
    int blocks = (DDIM * DDIM) / 256;
    transpose_f16_kernel<<<blocks, 256, 0, stream>>>(Wq, Wtq);
    transpose_f16_kernel<<<blocks, 256, 0, stream>>>(Wk, Wtk);
    transpose_f16_kernel<<<blocks, 256, 0, stream>>>(Wv, Wtv);
    transpose_f16_kernel<<<blocks, 256, 0, stream>>>(Wg, Wtg);
    transpose_f16_kernel<<<blocks, 256, 0, stream>>>(Wo, Wto);
  }
  // 3) projections (WMMA): 64x64 tiles = (M/64)*(512/64) = 2048 waves; 8/block
  {
    int blocks = (MTOT / 64) * (DDIM / 64) / 8;   // 256
    gemm_f16_kernel<<<blocks, 256, 0, stream>>>(xb, Wtq, bq, Qb, MTOT, 0);
    gemm_f16_kernel<<<blocks, 256, 0, stream>>>(xb, Wtk, bk, Kb, MTOT, 0);
    gemm_f16_kernel<<<blocks, 256, 0, stream>>>(xb, Wtv, bv, Vb, MTOT, 0);
    gemm_f16_kernel<<<blocks, 256, 0, stream>>>(xb, Wtg, bg, Gb, MTOT, 1); // sigmoid
  }
  // 4) recurrent scan (state in registers)
  {
    int blocks = BATCH * (DDIM / EB);   // 128
    scan_kernel<<<blocks, 256, 0, stream>>>(Qb, Kb, Vb, Gb, Oi);
  }
  // 5) output projection
  {
    int n = (int)MD;
    cvt_f16_kernel<<<(n + 255) / 256, 256, 0, stream>>>(Oi, Ob, n);
    int blocks = (MTOT / 64) * (DDIM / 64) / 8;
    gemm_f16_kernel<<<blocks, 256, 0, stream>>>(Ob, Wto, bo, out, MTOT, 0);
  }
}
